// Gene_Performer_attn_30648886624678
// MI455X (gfx1250) — compile-verified, matrix operand order checked by reading, not run on hardware
//
#include <hip/hip_runtime.h>
#include <math.h>

typedef __bf16 bf16_t;
typedef bf16_t bf16x8 __attribute__((ext_vector_type(8)));
typedef bf16_t v16bf  __attribute__((ext_vector_type(16)));
typedef float  v8f    __attribute__((ext_vector_type(8)));

#define SEQ   2048
#define EMB   128
#define HD    16
#define NH    8
#define RB    8
#define NF    128
#define DFOLD 1024            // RB*NF
#define ROWS  16384           // RB*SEQ
static const float RATIO = 0.08838834764831843f;  // 128^-0.5
static const float EPSF  = 1e-4f;

// ---- WMMA fragment load: A(16x32) / B(32x16 given as 16 rows x 32 K) bf16 ----
// lanes 0-15: row m=lane, K = [koff..koff+7] in v0..3, [16+koff..] in v4..7 (koff=0)
// lanes16-31: same rows, koff=8.  Two 16B loads per lane.
__device__ __forceinline__ v16bf load_frag(const bf16_t* base, int ld, int lane) {
  int row  = lane & 15;
  int koff = (lane >> 4) << 3;
  const bf16_t* p = base + (long)row * ld + koff;
  bf16x8 lo = *(const bf16x8*)(p);
  bf16x8 hi = *(const bf16x8*)(p + 16);
  return __builtin_shufflevector(lo, hi, 0,1,2,3,4,5,6,7,8,9,10,11,12,13,14,15);
}

// MODE 0: f32 store (nontemporal)   MODE 1: bf16 store
// MODE 2: +bias[col], bf16 store    MODE 3: *Dinv[row], permuted f32 store to final out
template<int MODE>
__global__ __launch_bounds__(128)
void gemm_wmma(const bf16_t* __restrict__ A, int lda, long sA1, long sA2,
               const bf16_t* __restrict__ Bt, int ldb, long sB1, long sB2,
               void* __restrict__ Cv, int ldc, long sC1, long sC2,
               int K, const float* __restrict__ extra, long sE1, int zshift)
{
  const int lane = threadIdx.x;
  const int wave = threadIdx.y;
  const int z  = blockIdx.z;
  const int zm = (1 << zshift) - 1;
  const long offA = (long)(z >> zshift) * sA1 + (long)(z & zm) * sA2;
  const long offB = (long)(z >> zshift) * sB1 + (long)(z & zm) * sB2;
  const long offC = (long)(z >> zshift) * sC1 + (long)(z & zm) * sC2;

  const int row0 = blockIdx.y * 128 + (wave >> 1) * 64;
  const int col0 = blockIdx.x * 128 + (wave & 1) * 64;
  const bf16_t* Ab = A + offA;
  const bf16_t* Bb = Bt + offB;

  v8f acc[4][4] = {};

  for (int k0 = 0; k0 < K; k0 += 32) {
    v16bf a[4], b[4];
    #pragma unroll
    for (int i = 0; i < 4; i++) a[i] = load_frag(Ab + (long)(row0 + i*16)*lda + k0, lda, lane);
    #pragma unroll
    for (int j = 0; j < 4; j++) b[j] = load_frag(Bb + (long)(col0 + j*16)*ldb + k0, ldb, lane);
    #pragma unroll
    for (int i = 0; i < 4; i++)
      #pragma unroll
      for (int j = 0; j < 4; j++)
        acc[i][j] = __builtin_amdgcn_wmma_f32_16x16x32_bf16(
            false, a[i], false, b[j], (short)0, acc[i][j], false, false);
  }

  const int rAdd = (lane >> 4) << 3;   // D layout: vgpr g -> row g (+8 for upper lanes)
  const int lcol = lane & 15;
  #pragma unroll
  for (int i = 0; i < 4; i++) {
    #pragma unroll
    for (int j = 0; j < 4; j++) {
      #pragma unroll
      for (int g = 0; g < 8; g++) {
        int r = row0 + i*16 + g + rAdd;
        int c = col0 + j*16 + lcol;
        float v = acc[i][j][g];
        if (MODE == 0) {
          __builtin_nontemporal_store(v, &((float*)Cv)[offC + (long)r*ldc + c]);
        } else if (MODE == 1) {
          ((bf16_t*)Cv)[offC + (long)r*ldc + c] = (bf16_t)v;
        } else if (MODE == 2) {
          ((bf16_t*)Cv)[offC + (long)r*ldc + c] = (bf16_t)(v + extra[(long)z*sE1 + c]);
        } else {
          float s = extra[(long)z*sE1 + r];          // Dinv[h*2048+n]
          long dst = ((long)(c >> 4) * SEQ + r) * EMB + z*HD + (c & 15);
          __builtin_nontemporal_store(v * s, &((float*)Cv)[dst]);
        }
      }
    }
  }
}

// ---------------- auxiliary kernels ----------------
__global__ void k_cvt(const float* __restrict__ s, bf16_t* __restrict__ d, int n) {
  int i = blockIdx.x*256 + threadIdx.x;
  if (i < n) d[i] = (bf16_t)s[i];
}

__global__ void k_prep_w(const float* Wq, const float* Wk, const float* Wv, bf16_t* wb) {
  int i = blockIdx.x*256 + threadIdx.x;
  if (i >= 3*16384) return;
  const float* s = (i < 16384) ? Wq : (i < 32768 ? Wk : Wv);
  wb[i] = (bf16_t)s[i & 16383];
}

__global__ void k_prep_small(const float* proj, const float* bq, const float* bk,
                             const float* bv, bf16_t* projPad, float* biases) {
  int i = blockIdx.x*256 + threadIdx.x;
  if (i < 128*32) {                         // proj padded to K=32, normalizer=0.5 folded in
    int j = i >> 5, kk = i & 31;
    projPad[i] = (bf16_t)(kk < 16 ? 0.5f * proj[j*16 + kk] : 0.0f);
  } else if (i < 128*32 + 384) {
    int t = i - 128*32;
    const float* b = t < 128 ? bq : (t < 256 ? bk : bv);
    biases[t] = b[t & 127];
  }
}

// q: per-(r,h,n) row max over 128 feats, exp, in place (folded layout slice)
__global__ void k_qfeat(bf16_t* __restrict__ qd, const bf16_t* __restrict__ yqb) {
  int tid = blockIdx.x*256 + threadIdx.x;    // 131072 = h*2048*8
  int r = tid & 7, n = (tid >> 3) & 2047, h = tid >> 14;
  bf16_t* slice = qd + (((long)(h*SEQ + n)) << 10) + (r << 7);
  const bf16_t* y = yqb + (((long)(r*SEQ + n)) << 7) + (h << 4);
  float diag = 0.f;
  #pragma unroll
  for (int e = 0; e < 16; e++) { float t = (float)y[e]; diag += t*t; }
  diag *= 0.125f;                             // 0.5 * normalizer^2
  float m = -1e30f;
  for (int j = 0; j < 128; j++) m = fmaxf(m, (float)slice[j]);
  float b = diag + m;
  for (int j = 0; j < 128; j++)
    slice[j] = (bf16_t)(RATIO * (expf((float)slice[j] - b) + EPSF));
}

__global__ void k_kmax(const bf16_t* __restrict__ kd, float* __restrict__ kmaxb) {
  __shared__ float red[256];
  int z = blockIdx.x, r = z >> 3, h = z & 7;
  const bf16_t* base = kd + ((long)h << 21) + (r << 7);
  float m = -1e30f;
  for (int it = 0; it < 1024; it++) {
    int flat = threadIdx.x + it*256;
    m = fmaxf(m, (float)base[((long)(flat >> 7) << 10) + (flat & 127)]);
  }
  red[threadIdx.x] = m; __syncthreads();
  for (int s = 128; s > 0; s >>= 1) {
    if (threadIdx.x < s) red[threadIdx.x] = fmaxf(red[threadIdx.x], red[threadIdx.x+s]);
    __syncthreads();
  }
  if (threadIdx.x == 0) kmaxb[z] = red[0];
}

__global__ void k_kfeat(bf16_t* __restrict__ kd, const bf16_t* __restrict__ ykb,
                        const float* __restrict__ kmaxb) {
  int tid = blockIdx.x*256 + threadIdx.x;
  int r = tid & 7, n = (tid >> 3) & 2047, h = tid >> 14;
  bf16_t* slice = kd + (((long)(h*SEQ + n)) << 10) + (r << 7);
  const bf16_t* y = ykb + (((long)(r*SEQ + n)) << 7) + (h << 4);
  float diag = 0.f;
  #pragma unroll
  for (int e = 0; e < 16; e++) { float t = (float)y[e]; diag += t*t; }
  float b = diag * 0.125f + kmaxb[r*8 + h];
  for (int j = 0; j < 128; j++)
    slice[j] = (bf16_t)(RATIO * (expf((float)slice[j] - b) + EPSF));
}

__global__ void k_transpose_kfT(const bf16_t* __restrict__ kf, bf16_t* __restrict__ kfT) {
  __shared__ bf16_t tile[32][33];
  int h = blockIdx.z, n0 = blockIdx.x*32, d0 = blockIdx.y*32;
  const bf16_t* src = kf  + ((long)h << 21);
  bf16_t*       dst = kfT + ((long)h << 21);
  int tx = threadIdx.x, ty = threadIdx.y;
  for (int i = 0; i < 4; i++)
    tile[ty + i*8][tx] = src[((long)(n0 + ty + i*8) << 10) + d0 + tx];
  __syncthreads();
  for (int i = 0; i < 4; i++)
    dst[((long)(d0 + ty + i*8) << 11) + n0 + tx] = tile[tx][ty + i*8];
}

__global__ void k_vfT(const bf16_t* __restrict__ yvb, bf16_t* __restrict__ vfT) {
  int idx = blockIdx.x*256 + threadIdx.x;     // 2,097,152
  int c = idx & 2047, d = (idx >> 11) & 127, h = idx >> 18;
  vfT[idx] = yvb[(((long)(d >> 4) * SEQ + c) << 7) + (h << 4) + (d & 15)];
}

__global__ void k_ksum(const bf16_t* __restrict__ kf, float* __restrict__ ksum) {
  int d = blockIdx.x*256 + threadIdx.x;       // 8192
  int h = d >> 10, dd = d & 1023;
  const bf16_t* base = kf + ((long)h << 21) + dd;
  float s = 0.f;
  for (int n = 0; n < SEQ; n++) s += (float)base[(long)n << 10];
  ksum[d] = s;
}

__global__ void k_dinv(const bf16_t* __restrict__ qf, const float* __restrict__ ksum,
                       float* __restrict__ dinv) {
  int wid  = blockIdx.x*8 + (threadIdx.x >> 5);   // 16384 rows
  int lane = threadIdx.x & 31;
  int h = wid >> 11;
  const bf16_t* row = qf + ((long)wid << 10);
  const float*  ks  = ksum + (h << 10);
  float s = 0.f;
  for (int i = lane; i < DFOLD; i += 32) s += (float)row[i] * ks[i];
  for (int m = 16; m > 0; m >>= 1) s += __shfl_xor(s, m, 32);
  if (lane == 0) dinv[wid] = 1.0f / s;
}

extern "C" void kernel_launch(void* const* d_in, const int* in_sizes, int n_in,
                              void* d_out, int out_size, void* d_ws, size_t ws_size,
                              hipStream_t stream) {
  const float* x  = (const float*)d_in[0];
  const float* Wq = (const float*)d_in[1];
  const float* bq = (const float*)d_in[2];
  const float* Wk = (const float*)d_in[3];
  const float* bk = (const float*)d_in[4];
  const float* Wv = (const float*)d_in[5];
  const float* bv = (const float*)d_in[6];
  const float* pj = (const float*)d_in[7];

  char* ws = (char*)d_ws; size_t off = 0;
  auto alloc = [&](size_t bytes) { void* p = ws + off; off = (off + bytes + 255) & ~(size_t)255; return p; };

  bf16_t* xb      = (bf16_t*)alloc(2097152ull*2);
  bf16_t* wb      = (bf16_t*)alloc(49152ull*2);
  float*  biases  = (float*) alloc(384ull*4);
  bf16_t* projPad = (bf16_t*)alloc(4096ull*2);
  bf16_t* yb      = (bf16_t*)alloc(3ull*2097152*2);
  bf16_t* qf      = (bf16_t*)alloc(16777216ull*2);   // qd -> qf in place
  bf16_t* kf      = (bf16_t*)alloc(16777216ull*2);   // kd -> kf in place
  bf16_t* kfT     = (bf16_t*)alloc(16777216ull*2);
  bf16_t* vfT     = (bf16_t*)alloc(2097152ull*2);
  bf16_t* ctxTb   = (bf16_t*)alloc(1048576ull*2);
  float*  kmaxb   = (float*) alloc(64ull*4);
  float*  ksum    = (float*) alloc(8192ull*4);
  float*  dinv    = (float*) alloc(16384ull*4);

  bf16_t* yqb = yb;
  bf16_t* ykb = yb + 2097152;
  bf16_t* yvb = yb + 2ull*2097152;
  float* outp  = (float*)d_out;
  float* attnp = outp + 2097152;

  dim3 gblk(32, 4, 1);

  k_cvt<<<8192,256,0,stream>>>(x, xb, 2097152);
  k_prep_w<<<192,256,0,stream>>>(Wq, Wk, Wv, wb);
  k_prep_small<<<18,256,0,stream>>>(pj, bq, bk, bv, projPad, biases);

  // projections: yb[z] = xb @ Wb[z]^T + b[z]   (M=16384, N=128, K=128), z=0..2
  gemm_wmma<2><<<dim3(1,128,3), gblk, 0, stream>>>(
      xb, 128, 0, 0,   wb, 128, 16384, 0,
      yb, 128, 2097152, 0,   128, biases, 128, 0);

  // feature GEMMs into folded (h,n,r*128+j) layout: z=(r<<3)|h, K=32 (padded)
  gemm_wmma<1><<<dim3(1,16,64), gblk, 0, stream>>>(
      yqb, 128, 262144, 16,   projPad, 32, 0, 0,
      qf, 1024, 128, 2097152,   32, nullptr, 0, 3);
  gemm_wmma<1><<<dim3(1,16,64), gblk, 0, stream>>>(
      ykb, 128, 262144, 16,   projPad, 32, 0, 0,
      kf, 1024, 128, 2097152,   32, nullptr, 0, 3);

  k_qfeat<<<512,256,0,stream>>>(qf, yqb);
  k_kmax <<<64, 256,0,stream>>>(kf, kmaxb);
  k_kfeat<<<512,256,0,stream>>>(kf, ykb, kmaxb);
  k_transpose_kfT<<<dim3(64,32,8), dim3(32,8), 0, stream>>>(kf, kfT);
  k_vfT <<<8192,256,0,stream>>>(yvb, vfT);
  k_ksum<<<32,  256,0,stream>>>(kf, ksum);
  k_dinv<<<2048,256,0,stream>>>(qf, ksum, dinv);

  // attn[h] = qf @ kf^T  (2048x2048, K=1024) -> f32 nontemporal
  gemm_wmma<0><<<dim3(16,16,8), gblk, 0, stream>>>(
      qf, 1024, 2097152, 0,   kf, 1024, 2097152, 0,
      attnp, 2048, 4194304, 0,   1024, nullptr, 0, 0);

  // ctxT[h] = vfT @ kf (Bt=kfT): (128x1024, K=2048) -> bf16
  gemm_wmma<1><<<dim3(8,1,8), gblk, 0, stream>>>(
      vfT, 2048, 262144, 0,   kfT, 2048, 2097152, 0,
      ctxTb, 1024, 131072, 0,   2048, nullptr, 0, 0);

  // out[h] = (qf @ ctx) * Dinv, permuted into (R,C,E): (2048x128, K=1024)
  gemm_wmma<3><<<dim3(1,16,8), gblk, 0, stream>>>(
      qf, 1024, 2097152, 0,   ctxTb, 1024, 131072, 0,
      outp, 0, 0, 0,   1024, dinv, 2048, 0);
}
